// RouteNet_53145925321278
// MI455X (gfx1250) — matrix operand order repaired and checked by reading, not compiled, not run on hardware
//
#include <hip/hip_runtime.h>

// ---------------------------------------------------------------------------
// RouteNet fused forward for MI455X (gfx1250, wave32, WMMA).
//
// B=65536, N_IN=784, D=10, NB=10, N_OUT=10.
// Pipeline (all bf16 WMMA, f32 accumulate):
//   prep_gate  : fp32-exact gate mask from batch row 0; bakes mask/any_open
//                into bf16 W2T[128n][128k], W3T[16n][128k], fused biases.
//   prep_w1    : W_in -> bf16 W1T[128n][800k] (K padded 784->800, N 100->128;
//                rows 112..127 are staging padding only, never multiplied).
//   main       : per 128-row batch tile: GEMM1 -> relu -> GEMM2 -> relu ->
//                GEMM3 -> log_softmax; one NT pass over x, logits stored NT.
//
// d_ws usage: 204800 (W1T) + 32768 (W2T) + 4096 (W3T) + 960 (biases) bytes.
// ---------------------------------------------------------------------------

typedef __bf16 v16bf __attribute__((ext_vector_type(16)));
typedef float  v8f   __attribute__((ext_vector_type(8)));
typedef float  v4f   __attribute__((ext_vector_type(4)));

union Frag { uint4 q[2]; v16bf v; };

// Software round-to-nearest-even f32 -> bf16 (used in prep kernels only).
__device__ __forceinline__ unsigned short f2bf(float f) {
  unsigned int u = __float_as_uint(f);
  u += 0x7FFFu + ((u >> 16) & 1u);
  return (unsigned short)(u >> 16);
}

// Hardware packed f32x2 -> bf16x2 (clang builtin not declared for gfx1250
// on this toolchain; probe-verified the instruction assembles).
__device__ __forceinline__ unsigned int pack_bf16(float a, float b) {
  unsigned int r;
  asm("v_cvt_pk_bf16_f32 %0, %1, %2" : "=v"(r) : "v"(a), "v"(b));
  return r;
}

// Hardware f32 -> bf16 in the low 16 bits (for ds_store_b16).
__device__ __forceinline__ unsigned short f2bf_hw(float f) {
  unsigned int r;
  asm("v_cvt_pk_bf16_f32 %0, %1, %2" : "=v"(r) : "v"(f), "v"(f));
  return (unsigned short)r;
}

__device__ __forceinline__ v8f vzero8() {
  v8f z;
#pragma unroll
  for (int j = 0; j < 8; ++j) z[j] = 0.0f;
  return z;
}

// ---------------------------------------------------------------------------
// prep_w1: W_in [10][784][10] f32 -> W1T [n=128][k=800] bf16 (zero padded)
// ---------------------------------------------------------------------------
__global__ void prep_w1(const float* __restrict__ W_in,
                        unsigned short* __restrict__ W1T) {
  int i = blockIdx.x * 256 + threadIdx.x;
  if (i >= 128 * 800) return;
  int n = i / 800, k = i % 800;
  float v = 0.0f;
  if (n < 100 && k < 784) {
    int s = n / 10, d = n % 10;
    v = W_in[(s * 784 + k) * 10 + d];
  }
  W1T[i] = f2bf(v);
}

// ---------------------------------------------------------------------------
// prep_gate: fp32-exact mask from batch row 0, then build masked bf16 weights.
// Single block, 256 threads.
// ---------------------------------------------------------------------------
__global__ void prep_gate(const float* __restrict__ x,
                          const float* __restrict__ W_in,
                          const float* __restrict__ b_in,
                          const float* __restrict__ W_gate,
                          const float* __restrict__ b_gate,
                          const float* __restrict__ W_data,
                          const float* __restrict__ b_data,
                          const float* __restrict__ W_out,
                          const float* __restrict__ b_out,
                          unsigned short* __restrict__ W2T,
                          unsigned short* __restrict__ W3T,
                          float* __restrict__ bias1,
                          float* __restrict__ bias2,
                          float* __restrict__ bias3,
                          float* __restrict__ prob_out) {
  __shared__ float a0[100];     // relu(acts0) for batch row 0, [s][d]
  __shared__ float mk[100];     // gate mask, [s][t]
  __shared__ float anyT[10];    // any incoming gate open, per target
  int tid = threadIdx.x;

  if (tid < 100) {              // acts0 row0: fp32-exact
    int s = tid / 10, d = tid % 10;
    float acc = b_in[tid];
    const float* wp = W_in + s * 7840 + d;
    for (int k = 0; k < 784; ++k) acc += x[k] * wp[k * 10];
    a0[tid] = acc > 0.0f ? acc : 0.0f;
  }
  __syncthreads();
  if (tid < 100) {              // gate[s][t]
    int s = tid / 10;
    float g = b_gate[tid];
    const float* wg = W_gate + tid * 10;
    for (int d = 0; d < 10; ++d) g += a0[s * 10 + d] * wg[d];
    mk[tid] = g > 0.0f ? 1.0f : 0.0f;
  }
  __syncthreads();
  if (tid < 10) {
    float s = 0.0f;
    for (int ss = 0; ss < 10; ++ss) s += mk[ss * 10 + tid];
    anyT[tid] = s > 0.0f ? 1.0f : 0.0f;
  }
  if (tid == 0) {
    float s = 0.0f;
    for (int i = 0; i < 100; ++i) s += mk[i];
    *prob_out = s / 20.0f;      // prob_open_gate (N_HIDD_BANKS = 20)
  }
  __syncthreads();

  // W2T[n=t*10+e pad 128][k=s*10+d pad 128] = mask[s][t] * W_data[s][t][d][e]
  for (int i = tid; i < 128 * 128; i += 256) {
    int n = i >> 7, k = i & 127;
    float v = 0.0f;
    if (n < 100 && k < 100) {
      int t = n / 10, e = n % 10, s = k / 10, d = k % 10;
      v = mk[s * 10 + t] * W_data[((s * 10 + t) * 10 + d) * 10 + e];
    }
    W2T[i] = f2bf(v);
  }
  // W3T[n=o pad 16][k=t*10+e pad 128] = any_open[t] * W_out[t][e][o]
  for (int i = tid; i < 16 * 128; i += 256) {
    int n = i >> 7, k = i & 127;
    float v = 0.0f;
    if (n < 10 && k < 100) {
      int t = k / 10, e = k % 10;
      v = anyT[t] * W_out[(t * 10 + e) * 10 + n];
    }
    W3T[i] = f2bf(v);
  }
  if (tid < 112) {
    bias1[tid] = (tid < 100) ? b_in[tid] : 0.0f;
    float b2 = 0.0f;
    if (tid < 100) {
      int t = tid / 10, e = tid % 10;
      for (int s = 0; s < 10; ++s)
        b2 += mk[s * 10 + t] * b_data[(s * 10 + t) * 10 + e];
    }
    bias2[tid] = b2;
  }
  if (tid < 16) {
    float b3 = 0.0f;
    if (tid < 10)
      for (int t = 0; t < 10; ++t) b3 += anyT[t] * b_out[t * 10 + tid];
    bias3[tid] = b3;
  }
}

// ---------------------------------------------------------------------------
// Fused main kernel. 256 threads = 8 waves; wave w owns batch rows
// [wgBase + 16w, +16). LDS pitches (40 / 136 halfwords) keep the 16-byte
// ds_load_b128 fragment gathers on distinct banks. All staging loops are
// counted (trip 4 / trip 2, no residues) so no exec masking is needed, and
// loads are issued as a batch before the converts so s_wait_loadcnt is
// staged instead of full-drain.
// ---------------------------------------------------------------------------
__global__ __launch_bounds__(256, 2) void routenet_main(
    const float* __restrict__ x,
    const unsigned short* __restrict__ W1T,   // [128][800]
    const unsigned short* __restrict__ W2T,   // [128][128]
    const unsigned short* __restrict__ W3T,   // [16][128]
    const float* __restrict__ bias1,
    const float* __restrict__ bias2,
    const float* __restrict__ bias3,
    float* __restrict__ out) {
  __shared__ unsigned short bufA[128 * 40];   // x tile, bf16, K-chunk of 32
  __shared__ unsigned short bufB[128 * 40];   // weight K-chunk of 32
  __shared__ unsigned short acts[128 * 136];  // acts0 then acts1, [row][128pad]
  __shared__ float sb1[112], sb2[112], sb3[16];

  const int tid   = threadIdx.x;
  const int lane  = tid & 31;
  const int wave  = tid >> 5;
  const int WR    = wave * 16;                // wave's local row base
  const int m     = lane & 15;                // col-in-tile / row-in-A
  const int hi    = lane >> 4;                // half-wave select
  const int g     = hi << 3;                  // unused; A-frag also uses kb
  const int kb    = hi << 4;                  // B-frag K offset (0 / 16)
  const int mOff  = hi << 3;                  // C row offset (0 / 8)
  const long wgBase = (long)blockIdx.x * 128;

  // Per-thread staging geometry (constant across K-chunks):
  const int xrow  = tid >> 3;                 // 0..31, +32 per it
  const int xc4   = tid & 7;                  // float4 slot within 32-K chunk
  const float* xbase = x + (size_t)(wgBase + xrow) * 784 + xc4 * 4;
  const int aoff  = xrow * 40 + xc4 * 4;      // bufA halfword offset
  const int wn    = tid >> 2;                 // 0..63, +64 per it
  const int wq    = tid & 3;                  // uint4 slot within 32-K chunk
  const int boff  = wn * 40 + wq * 8;         // bufB halfword offset

  if (tid < 112) { sb1[tid] = bias1[tid]; sb2[tid] = bias2[tid]; }
  if (tid < 16)  { sb3[tid] = bias3[tid]; }

  // ---------------- GEMM1: acts0 = relu(x @ W1 + b1), K = 800 ----------------
  v8f acc[7];
#pragma unroll
  for (int c = 0; c < 7; ++c) acc[c] = vzero8();

  for (int kc = 0; kc < 25; ++kc) {
    __syncthreads();                          // protect prior-iter LDS reads
    // x chunk: 128 rows x 32 K (f32 -> bf16), non-temporal (streamed once)
    if (kc < 24) {                            // uniform branch: full chunks
      v4f vv[4];
#pragma unroll
      for (int it = 0; it < 4; ++it)          // 4 NT b128 loads in flight
        vv[it] = __builtin_nontemporal_load(
            (const v4f*)(xbase + kc * 32 + it * (32 * 784)));
#pragma unroll
      for (int it = 0; it < 4; ++it) {
        uint2 p;
        p.x = pack_bf16(vv[it].x, vv[it].y);  // v_cvt_pk_bf16_f32
        p.y = pack_bf16(vv[it].z, vv[it].w);
        *(uint2*)&bufA[aoff + it * 1280] = p;
      }
    } else {                                  // tail chunk: k 768..799
      v4f vv[4];
#pragma unroll
      for (int it = 0; it < 4; ++it) {
        vv[it] = (v4f){0.0f, 0.0f, 0.0f, 0.0f};
        if (xc4 < 4)                          // k = 768 + xc4*4 < 784
          vv[it] = __builtin_nontemporal_load(
              (const v4f*)(xbase + 768 + it * (32 * 784)));
      }
#pragma unroll
      for (int it = 0; it < 4; ++it) {
        uint2 p;
        p.x = pack_bf16(vv[it].x, vv[it].y);
        p.y = pack_bf16(vv[it].z, vv[it].w);
        *(uint2*)&bufA[aoff + it * 1280] = p;
      }
    }
    // W1T chunk: 128 rows x 32 K, two full unguarded trips
#pragma unroll
    for (int it = 0; it < 2; ++it) {
      uint4 w = *(const uint4*)(W1T + (wn + it * 64) * 800 + kc * 32 + wq * 8);
      *(uint4*)&bufB[boff + it * 2560] = w;
    }
    __syncthreads();

    Frag a;                                    // 16x32 bf16 A, k = 16*hi + s (contiguous)
    a.q[0] = *(const uint4*)&bufA[(WR + m) * 40 + kb];
    a.q[1] = *(const uint4*)&bufA[(WR + m) * 40 + kb + 8];
#pragma unroll
    for (int c = 0; c < 7; ++c) {
      Frag b;                                  // 32x16 bf16 B, same contiguous k order as A
      int n = c * 16 + m;
      b.q[0] = *(const uint4*)&bufB[n * 40 + kb];
      b.q[1] = *(const uint4*)&bufB[n * 40 + kb + 8];
      acc[c] = __builtin_amdgcn_wmma_f32_16x16x32_bf16(
          false, a.v, false, b.v, (short)0, acc[c], false, false);
    }
  }

  // relu + bias -> acts0 (bf16) in LDS, zero the K-pad columns 112..127
#pragma unroll
  for (int c = 0; c < 7; ++c) {
#pragma unroll
    for (int j = 0; j < 8; ++j) {
      int mm = j + mOff, n = c * 16 + m;
      float v = acc[c][j] + sb1[n];
      acts[(WR + mm) * 136 + n] = f2bf_hw(v > 0.0f ? v : 0.0f);
    }
  }
#pragma unroll
  for (int j = 0; j < 8; ++j)
    acts[(WR + j + mOff) * 136 + 112 + m] = 0;

  // ---------------- GEMM2: acts1 = relu(acts0 @ W2 + b2), K = 128 -----------
  v8f acc2[7];
#pragma unroll
  for (int c = 0; c < 7; ++c) acc2[c] = vzero8();

  for (int kc = 0; kc < 4; ++kc) {
    __syncthreads();
#pragma unroll
    for (int it = 0; it < 2; ++it) {
      uint4 w = *(const uint4*)(W2T + (wn + it * 64) * 128 + kc * 32 + wq * 8);
      *(uint4*)&bufB[boff + it * 2560] = w;
    }
    __syncthreads();

    Frag a;
    a.q[0] = *(const uint4*)&acts[(WR + m) * 136 + kc * 32 + kb];
    a.q[1] = *(const uint4*)&acts[(WR + m) * 136 + kc * 32 + kb + 8];
#pragma unroll
    for (int c = 0; c < 7; ++c) {
      Frag b;
      int n = c * 16 + m;
      b.q[0] = *(const uint4*)&bufB[n * 40 + kb];
      b.q[1] = *(const uint4*)&bufB[n * 40 + kb + 8];
      acc2[c] = __builtin_amdgcn_wmma_f32_16x16x32_bf16(
          false, a.v, false, b.v, (short)0, acc2[c], false, false);
    }
  }

  // relu + bias -> acts1 overwrites acts (wave only touches its own 16 rows)
#pragma unroll
  for (int c = 0; c < 7; ++c) {
#pragma unroll
    for (int j = 0; j < 8; ++j) {
      int mm = j + mOff, n = c * 16 + m;
      float v = acc2[c][j] + sb2[n];
      acts[(WR + mm) * 136 + n] = f2bf_hw(v > 0.0f ? v : 0.0f);
    }
  }
#pragma unroll
  for (int j = 0; j < 8; ++j)
    acts[(WR + j + mOff) * 136 + 112 + m] = 0;

  // ---------------- GEMM3: logits = acts1 @ W3 + b3, K = 128, N = 16 --------
  v8f acc3 = vzero8();
  for (int kc = 0; kc < 4; ++kc) {
    __syncthreads();
    if (tid < 64) {                           // 16 rows x 4 quads
      uint4 w = *(const uint4*)(W3T + wn * 128 + kc * 32 + wq * 8);
      *(uint4*)&bufB[boff] = w;
    }
    __syncthreads();

    Frag a;
    a.q[0] = *(const uint4*)&acts[(WR + m) * 136 + kc * 32 + kb];
    a.q[1] = *(const uint4*)&acts[(WR + m) * 136 + kc * 32 + kb + 8];
    Frag b;
    b.q[0] = *(const uint4*)&bufB[m * 40 + kb];
    b.q[1] = *(const uint4*)&bufB[m * 40 + kb + 8];
    acc3 = __builtin_amdgcn_wmma_f32_16x16x32_bf16(
        false, a.v, false, b.v, (short)0, acc3, false, false);
  }

  // ---------------- log_softmax over 10 valid columns (lane-shuffle) --------
#pragma unroll
  for (int j = 0; j < 8; ++j) {
    int mm = j + mOff;
    long rowG = wgBase + WR + mm;
    float v = acc3[j] + sb3[m];
    float mv = (m < 10) ? v : -3.0e38f;
#pragma unroll
    for (int sh = 1; sh < 16; sh <<= 1) mv = fmaxf(mv, __shfl_xor(mv, sh));
    float p = (m < 10) ? __expf(v - mv) : 0.0f;
#pragma unroll
    for (int sh = 1; sh < 16; sh <<= 1) p += __shfl_xor(p, sh);
    if (m < 10)
      __builtin_nontemporal_store(v - mv - __logf(p), &out[rowG * 10 + m]);
  }
}

// ---------------------------------------------------------------------------
extern "C" void kernel_launch(void* const* d_in, const int* in_sizes, int n_in,
                              void* d_out, int out_size, void* d_ws,
                              size_t ws_size, hipStream_t stream) {
  (void)in_sizes; (void)n_in; (void)out_size; (void)ws_size;
  const float* x      = (const float*)d_in[0];
  const float* W_in   = (const float*)d_in[1];
  const float* b_in   = (const float*)d_in[2];
  const float* W_gate = (const float*)d_in[3];
  const float* b_gate = (const float*)d_in[4];
  const float* W_data = (const float*)d_in[5];
  const float* b_data = (const float*)d_in[6];
  const float* W_out  = (const float*)d_in[7];
  const float* b_out  = (const float*)d_in[8];
  float* out = (float*)d_out;

  char* ws = (char*)d_ws;                     // needs >= 242,624 bytes
  unsigned short* W1T = (unsigned short*)(ws);                  // 204,800 B
  unsigned short* W2T = (unsigned short*)(ws + 204800);         //  32,768 B
  unsigned short* W3T = (unsigned short*)(ws + 204800 + 32768); //   4,096 B
  float* bias1 = (float*)(ws + 204800 + 32768 + 4096);
  float* bias2 = bias1 + 112;
  float* bias3 = bias2 + 112;
  float* prob  = out + (size_t)65536 * 10;    // scalar output #2

  prep_w1<<<(128 * 800 + 255) / 256, 256, 0, stream>>>(W_in, W1T);
  prep_gate<<<1, 256, 0, stream>>>(x, W_in, b_in, W_gate, b_gate, W_data,
                                   b_data, W_out, b_out, W2T, W3T, bias1,
                                   bias2, bias3, prob);
  routenet_main<<<65536 / 128, 256, 0, stream>>>(x, W1T, W2T, W3T, bias1,
                                                 bias2, bias3, out);
}
